// CausalHierarchicalMemoryLM_72971494359525
// MI455X (gfx1250) — matrix operand order repaired and checked by reading, not compiled, not run on hardware
//
#include <hip/hip_runtime.h>
#include <hip/hip_bf16.h>

// ---------------------------------------------------------------------------
// CausalHierarchicalMemoryLM for MI455X (gfx1250, wave32, WMMA)
//
// Vocab projection (2047x512x32000, ~67 GFLOP) is L2-bound, not compute-bound:
// E16 = 32MB lives in the 192MB L2. A 64x64 register-blocked wave tile plus
// LDS staging of the 64-row A supertile (async-to-LDS, ASYNCcnt) cuts L2
// reads from ~5GB to ~1.3GB. Scan stays sequential in one workgroup with an
// incrementally-maintained KR = M @ Vr cache. Softmax/LN/attention in f32.
// ---------------------------------------------------------------------------

typedef _Float16 v8h  __attribute__((ext_vector_type(8)));
typedef _Float16 v16h __attribute__((ext_vector_type(16)));
typedef float    v8f  __attribute__((ext_vector_type(8)));

#define NSEQ   2047
#define NPAD   2048
#define DIMX   512
#define RANKX  64
#define VOCABX 32000
#define SLOT0  256
#define SLOT1  64
#define SLOT2  16

#if __has_builtin(__builtin_amdgcn_global_load_async_to_lds_b128)
#define HAVE_ASYNC_LDS 1
#endif

// Async-to-LDS builtin takes pointers to 16B int vectors in AS1 (global) and
// AS3 (LDS) — per the hipcc diagnostic from the compile probe.
typedef int v4i_t __attribute__((vector_size(16)));
typedef __attribute__((address_space(1))) v4i_t as1_v4i;
typedef __attribute__((address_space(3))) v4i_t as3_v4i;

// ---------------------------------------------------------------------------
// Embedding gather: s[t][d] = E[ids[t+1]][d]
// ---------------------------------------------------------------------------
__global__ void embed_k(const int* __restrict__ ids, const float* __restrict__ E,
                        float* __restrict__ s) {
  int i = blockIdx.x * blockDim.x + threadIdx.x;
  if (i >= NSEQ * DIMX) return;
  int t = i >> 9, d = i & 511;
  s[i] = E[(size_t)ids[t + 1] * DIMX + d];
}

// f32 -> f16 convert with zero-padding beyond nsrc
__global__ void cvt_k(const float* __restrict__ src, _Float16* __restrict__ dst,
                      int n, int nsrc) {
  int i = blockIdx.x * blockDim.x + threadIdx.x;
  if (i < n) dst[i] = (i < nsrc) ? (_Float16)src[i] : (_Float16)0.f;
}

// W: (2, DIM, RANK) f32  ->  dst: (2, RANK, DIM) f16   (B^T layout for GEMM)
__global__ void tw_k(const float* __restrict__ W, _Float16* __restrict__ dst) {
  int i = blockIdx.x * blockDim.x + threadIdx.x;
  if (i >= 2 * RANKX * DIMX) return;
  int p = i >> 15;
  int rem = i & 32767;
  int r = rem >> 9;
  int d = rem & 511;
  dst[i] = (_Float16)W[((size_t)p * DIMX + d) * RANKX + r];
}

// ---------------------------------------------------------------------------
// Small WMMA GEMM: C[M x Ncol] = A[Mpad x K] * Bt[Ncol x K]^T (16x64 per wave)
// Used for the Wproj / Wq / Wk projections (keeps launch-grain parallelism).
// ---------------------------------------------------------------------------
__global__ __launch_bounds__(256) void gemm_k(const _Float16* __restrict__ A,
                                              const _Float16* __restrict__ Bt,
                                              float* __restrict__ C,
                                              int Mreal, int K, int Ncol,
                                              int tilesN, int tilesM) {
  const int wave = (blockIdx.x * 256 + threadIdx.x) >> 5;
  const int lane = threadIdx.x & 31;
  const int tm = wave / tilesN;
  const int tn = wave % tilesN;
  if (tm >= tilesM) return;  // wave-uniform

  const int m = lane & 15;
  const int half = lane >> 4;
  const _Float16* arow = A + (size_t)(tm * 16 + m) * K;

  v8f acc[4];
  v8f zero = {0.f, 0.f, 0.f, 0.f, 0.f, 0.f, 0.f, 0.f};
#pragma unroll
  for (int f = 0; f < 4; ++f) acc[f] = zero;

  for (int k0 = 0; k0 < K; k0 += 32) {
    v8h alo = *(const v8h*)(arow + k0 + half * 16);
    v8h ahi = *(const v8h*)(arow + k0 + 8 + half * 16);
    v16h af = __builtin_shufflevector(alo, ahi, 0, 1, 2, 3, 4, 5, 6, 7,
                                      8, 9, 10, 11, 12, 13, 14, 15);
#pragma unroll
    for (int f = 0; f < 4; ++f) {
      const _Float16* brow =
          Bt + (size_t)(tn * 64 + f * 16 + m) * K + k0 + half * 16;
      v16h bf = *(const v16h*)brow;
      acc[f] = __builtin_amdgcn_wmma_f32_16x16x32_f16(
          false, af, false, bf, (short)0, acc[f], false, false);
    }
  }

  const int rbase = tm * 16 + half * 8;
#pragma unroll
  for (int f = 0; f < 4; ++f) {
    int col = tn * 64 + f * 16 + m;
#pragma unroll
    for (int i = 0; i < 8; ++i) {
      int row = rbase + i;
      if (row < Mreal) C[(size_t)row * Ncol + col] = acc[f][i];
    }
  }
}

// ---------------------------------------------------------------------------
// Big WMMA GEMM for the vocab projection. K fixed at 512.
// Block = 128 threads (4 waves). Each wave: 64(M) x 64(N) register tile
// (16 v8f accumulators). The shared 64-row A supertile is staged in LDS once
// (async-to-LDS when available), row-padded to 520 halves so the 16 row
// readers of ds_load_b128 hit disjoint bank groups.
// ---------------------------------------------------------------------------
#define APITCH 520  // 512 + 8 halves pad: row stride 1040B -> conflict-free

__global__ __launch_bounds__(128) void gemm64_k(const _Float16* __restrict__ A,
                                                const _Float16* __restrict__ Bt,
                                                float* __restrict__ C,
                                                int Mreal, int Ncol, int ngrp) {
  __shared__ _Float16 As[64 * APITCH];

  const int stm = blockIdx.x / ngrp;            // M supertile (64 rows)
  const int tng = blockIdx.x % ngrp;
  const int tn = tng * 4 + (threadIdx.x >> 5);  // wave-uniform N tile
  const int lane = threadIdx.x & 31;
  const int m = lane & 15;
  const int half = lane >> 4;

  // ---- stage A supertile (64 x 512 f16 = 64KB contiguous) into LDS
  {
    const _Float16* ag = A + (size_t)stm * 64 * DIMX;
#pragma unroll
    for (int i = 0; i < 32; ++i) {
      int c = i * 128 + threadIdx.x;  // 4096 chunks of 8 halves (16B)
      int r = c >> 6, x = c & 63;
      const _Float16* gsrc = ag + r * DIMX + x * 8;
      _Float16* ldst = &As[r * APITCH + x * 8];
#ifdef HAVE_ASYNC_LDS
      __builtin_amdgcn_global_load_async_to_lds_b128((as1_v4i*)gsrc,
                                                     (as3_v4i*)ldst, 0, 0);
#else
      *(v8h*)ldst = *(const v8h*)gsrc;
#endif
    }
  }
#ifdef HAVE_ASYNC_LDS
#if __has_builtin(__builtin_amdgcn_s_wait_asynccnt)
  __builtin_amdgcn_s_wait_asynccnt(0);
#else
  asm volatile("s_wait_asynccnt 0" ::: "memory");
#endif
#endif
  __syncthreads();

  v8f acc[4][4];
  v8f zero = {0.f, 0.f, 0.f, 0.f, 0.f, 0.f, 0.f, 0.f};
#pragma unroll
  for (int am = 0; am < 4; ++am)
#pragma unroll
    for (int f = 0; f < 4; ++f) acc[am][f] = zero;

  for (int k0 = 0; k0 < DIMX; k0 += 32) {
    v16h bf[4];
#pragma unroll
    for (int f = 0; f < 4; ++f)
      bf[f] = *(const v16h*)(Bt + (size_t)(tn * 64 + f * 16 + m) * DIMX + k0 +
                             half * 16);
#pragma unroll
    for (int am = 0; am < 4; ++am) {
      const _Float16* ar = &As[(am * 16 + m) * APITCH + k0 + half * 16];
      v8h alo = *(const v8h*)ar;
      v8h ahi = *(const v8h*)(ar + 8);
      v16h af = __builtin_shufflevector(alo, ahi, 0, 1, 2, 3, 4, 5, 6, 7,
                                        8, 9, 10, 11, 12, 13, 14, 15);
#pragma unroll
      for (int f = 0; f < 4; ++f)
        acc[am][f] = __builtin_amdgcn_wmma_f32_16x16x32_f16(
            false, af, false, bf[f], (short)0, acc[am][f], false, false);
    }
  }

#pragma unroll
  for (int am = 0; am < 4; ++am) {
    int rb = stm * 64 + am * 16 + half * 8;
#pragma unroll
    for (int f = 0; f < 4; ++f) {
      int col = tn * 64 + f * 16 + m;
#pragma unroll
      for (int i = 0; i < 8; ++i) {
        int row = rb + i;
        if (row < Mreal) C[(size_t)row * Ncol + col] = acc[am][f][i];
      }
    }
  }
}

// ---------------------------------------------------------------------------
// LayerNorm over last dim (512). One block (256 threads) per row.
// ---------------------------------------------------------------------------
__global__ __launch_bounds__(256) void ln_k(const float* __restrict__ src,
                                            const float* __restrict__ g,
                                            const float* __restrict__ b,
                                            float* __restrict__ dst) {
  __shared__ float rs[256], rq[256];
  int n = blockIdx.x, t = threadIdx.x;
  float x0 = src[(size_t)n * DIMX + t];
  float x1 = src[(size_t)n * DIMX + 256 + t];
  rs[t] = x0 + x1;
  rq[t] = x0 * x0 + x1 * x1;
  __syncthreads();
  for (int off = 128; off > 0; off >>= 1) {
    if (t < off) { rs[t] += rs[t + off]; rq[t] += rq[t + off]; }
    __syncthreads();
  }
  float mean = rs[0] * (1.f / 512.f);
  float var = rq[0] * (1.f / 512.f) - mean * mean;
  float inv = rsqrtf(var + 1e-5f);
  dst[(size_t)n * DIMX + t] = (x0 - mean) * inv * g[t] + b[t];
  dst[(size_t)n * DIMX + 256 + t] = (x1 - mean) * inv * g[256 + t] + b[256 + t];
}

// ---------------------------------------------------------------------------
// Hierarchical memory scan (sequential over t). Single workgroup, 512 threads.
// KR_l = M_l @ Vr[l] kept incrementally (top-16 rows updated per gated step).
// Top-16 selection via order-independent packed u64 atomicMax in LDS.
// ---------------------------------------------------------------------------
__device__ __forceinline__ unsigned ford(float f) {
  unsigned u = __float_as_uint(f);
  return (u & 0x80000000u) ? ~u : (u | 0x80000000u);
}

__global__ __launch_bounds__(512) void scan_k(
    const float* __restrict__ s, const float* __restrict__ Ur,
    const float* __restrict__ Vr, const float* __restrict__ m00,
    const float* __restrict__ m01, const float* __restrict__ m02,
    float* __restrict__ M0, float* __restrict__ M1, float* __restrict__ M2,
    float* __restrict__ KR0, float* __restrict__ KR1, float* __restrict__ KR2,
    float* __restrict__ q, _Float16* __restrict__ qf16) {
  __shared__ float xs[512];
  __shared__ float rd[512];
  __shared__ float pq[512];
  __shared__ float pv[512];
  __shared__ float qr[64];
  __shared__ float xv[64];
  __shared__ float sc[256];
  __shared__ float wsm[16];
  __shared__ float vals[16];
  __shared__ int idxs[16];
  __shared__ unsigned long long best;

  const int tid = threadIdx.x;

  // ---- prologue: copy initial memories, zero f16 pad row, build KR = M@Vr
  for (int i = tid; i < SLOT0 * DIMX; i += 512) M0[i] = m00[i];
  for (int i = tid; i < SLOT1 * DIMX; i += 512) M1[i] = m01[i];
  for (int i = tid; i < SLOT2 * DIMX; i += 512) M2[i] = m02[i];
  qf16[(size_t)NSEQ * DIMX + tid] = (_Float16)0.f;
  __syncthreads();

  float* Ms[3] = {M0, M1, M2};
  float* KRs[3] = {KR0, KR1, KR2};
  const int Ss[3] = {SLOT0, SLOT1, SLOT2};

  for (int l = 0; l < 3; ++l) {
    const float* M = Ms[l];
    float* KR = KRs[l];
    const float* V = Vr + (size_t)l * DIMX * RANKX;
    for (int p = tid; p < Ss[l] * RANKX; p += 512) {
      int i = p >> 6, r = p & 63;
      float acc = 0.f;
      for (int d = 0; d < DIMX; ++d) acc += M[i * DIMX + d] * V[d * RANKX + r];
      KR[p] = acc;
    }
  }
  __syncthreads();

  const float scale = 0.125f;  // 1/sqrt(RANK)

  for (int t = 0; t < NSEQ; ++t) {
    xs[tid] = s[(size_t)t * DIMX + tid];
    __syncthreads();
    rd[tid] = xs[tid];  // each thread only ever touches rd[tid]

    for (int l = 0; l < 3; ++l) {
      float* M = Ms[l];
      float* KR = KRs[l];
      const int S = Ss[l];
      const float* U = Ur + (size_t)l * DIMX * RANKX;
      const float* V = Vr + (size_t)l * DIMX * RANKX;

      // qr = x @ Ur[l], xv = x @ Vr[l] : two-stage deterministic reduction
      {
        int r = tid & 63, seg = tid >> 6;
        float aq = 0.f, av = 0.f;
        for (int i = 0; i < 64; ++i) {
          float xval = xs[seg * 64 + i];
          aq += xval * U[(seg * 64 + i) * RANKX + r];
          av += xval * V[(seg * 64 + i) * RANKX + r];
        }
        pq[tid] = aq;
        pv[tid] = av;
      }
      __syncthreads();
      if (tid < 64) {
        float aq = 0.f, av = 0.f;
        for (int g = 0; g < 8; ++g) { aq += pq[g * 64 + tid]; av += pv[g * 64 + tid]; }
        qr[tid] = aq;
        xv[tid] = av;
      }
      __syncthreads();

      // sc[i] = scale * <qr, KR[i]>
      if (tid < S) {
        float acc = 0.f;
        for (int r = 0; r < RANKX; ++r) acc += qr[r] * KR[tid * RANKX + r];
        sc[tid] = acc * scale;
      }
      __syncthreads();

      // top-16 (value desc, index asc on ties)
      for (int k = 0; k < 16; ++k) {
        if (tid == 0) best = 0ull;
        __syncthreads();
        if (tid < S) {
          unsigned long long pack =
              ((unsigned long long)ford(sc[tid]) << 32) | (unsigned)(~tid);
          atomicMax(&best, pack);
        }
        __syncthreads();
        if (tid == 0) {
          int bi = (int)(~(unsigned)best);
          idxs[k] = bi;
          vals[k] = sc[bi];
          sc[bi] = -3.0e38f;
        }
        __syncthreads();
      }

      // softmax over the 16 selected scores (vals[0] is the max)
      if (tid == 0) {
        float mx = vals[0], sum = 0.f;
        for (int k = 0; k < 16; ++k) { float e = expf(vals[k] - mx); wsm[k] = e; sum += e; }
        float inv = 1.f / sum;
        for (int k = 0; k < 16; ++k) wsm[k] *= inv;
      }
      __syncthreads();

      const bool gate = (l == 0) || (l == 1 && (t & 3) == 0) || (l == 2 && (t & 15) == 0);

      // read += sum_k w_k * M[idx_k]  (pre-update values), then gated write
      {
        float acc = 0.f;
        float xd = xs[tid];
        for (int k = 0; k < 16; ++k) {
          float w = wsm[k];
          float* mp = &M[idxs[k] * DIMX + tid];
          float mv = *mp;
          acc += w * mv;
          if (gate) *mp = mv + w * xd;
        }
        rd[tid] += acc;
      }
      if (gate && tid < 64) {
        for (int k = 0; k < 16; ++k)
          KR[idxs[k] * RANKX + tid] += wsm[k] * xv[tid];
      }
      __syncthreads();
    }

    q[(size_t)t * DIMX + tid] = rd[tid];
    qf16[(size_t)t * DIMX + tid] = (_Float16)rd[tid];
    __syncthreads();
  }
}

// ---------------------------------------------------------------------------
// Signed sliding-window attention (WIN=64). One block (64 threads) per query.
// ---------------------------------------------------------------------------
__global__ __launch_bounds__(64) void attn_k(const float* __restrict__ h,
                                             const float* __restrict__ a,
                                             const float* __restrict__ kf,
                                             float* __restrict__ outp) {
  __shared__ float aL[64];
  __shared__ float wgt[64];
  __shared__ int idc[64];
  __shared__ float red[64];
  const int n = blockIdx.x;
  const int t = threadIdx.x;

  aL[t] = a[(size_t)n * RANKX + t];
  __syncthreads();

  int j = n - 63 + t;
  int valid = j >= 0;
  int jc = j < 0 ? 0 : j;
  idc[t] = jc;
  const float* kp = kf + (size_t)jc * RANKX;
  float sc_ = 0.f;
  for (int r = 0; r < RANKX; ++r) sc_ += aL[r] * kp[r];
  sc_ *= 0.125f;
  float logit = valid ? fabsf(sc_) : -1.0e9f;

  red[t] = logit;
  __syncthreads();
  for (int off = 32; off > 0; off >>= 1) {
    if (t < off) red[t] = fmaxf(red[t], red[t + off]);
    __syncthreads();
  }
  float mx = red[0];
  __syncthreads();
  float e = expf(logit - mx);
  red[t] = e;
  __syncthreads();
  for (int off = 32; off > 0; off >>= 1) {
    if (t < off) red[t] += red[t + off];
    __syncthreads();
  }
  float inv = 1.f / red[0];
  float sgn = (sc_ > 0.f) ? 1.f : (sc_ < 0.f ? -1.f : 0.f);
  wgt[t] = e * inv * sgn * (valid ? 1.f : 0.f);
  __syncthreads();

  float acc[8] = {0.f, 0.f, 0.f, 0.f, 0.f, 0.f, 0.f, 0.f};
  for (int w = 0; w < 64; ++w) {
    float wv = wgt[w];
    const float* hr = h + (size_t)idc[w] * DIMX;
#pragma unroll
    for (int i = 0; i < 8; ++i) acc[i] += wv * hr[i * 64 + t];
  }
#pragma unroll
  for (int i = 0; i < 8; ++i)
    outp[(size_t)n * DIMX + i * 64 + t] = h[(size_t)n * DIMX + i * 64 + t] + acc[i];
}

// ---------------------------------------------------------------------------
// Host orchestration
// ---------------------------------------------------------------------------
extern "C" void kernel_launch(void* const* d_in, const int* in_sizes, int n_in,
                              void* d_out, int out_size, void* d_ws, size_t ws_size,
                              hipStream_t stream) {
  const int* ids = (const int*)d_in[0];
  const float* E = (const float*)d_in[1];
  const float* Ur = (const float*)d_in[2];
  const float* Vr = (const float*)d_in[3];
  const float* m00 = (const float*)d_in[4];
  const float* m01 = (const float*)d_in[5];
  const float* m02 = (const float*)d_in[6];
  const float* Wproj = (const float*)d_in[7];
  const float* Wq = (const float*)d_in[8];
  const float* Wk = (const float*)d_in[9];
  const float* ln_in_g = (const float*)d_in[10];
  const float* ln_in_b = (const float*)d_in[11];
  const float* ln_pred_g = (const float*)d_in[12];
  const float* ln_pred_b = (const float*)d_in[13];
  const float* ln_out_g = (const float*)d_in[14];
  const float* ln_out_b = (const float*)d_in[15];
  float* out = (float*)d_out;

  char* w = (char*)d_ws;
  size_t off = 0;
  auto take = [&](size_t bytes) -> char* {
    char* p = w + off;
    off = (off + bytes + 255) & ~(size_t)255;
    return p;
  };

  float* s = (float*)take((size_t)NPAD * DIMX * 4);
  float* q = (float*)take((size_t)NPAD * DIMX * 4);
  _Float16* qf16 = (_Float16*)take((size_t)NPAD * DIMX * 2);
  float* tmp = (float*)take((size_t)NPAD * DIMX * 4);
  float* h = (float*)take((size_t)NPAD * DIMX * 4);
  _Float16* hf16 = (_Float16*)take((size_t)NPAD * DIMX * 2);
  float* abuf = (float*)take((size_t)NPAD * RANKX * 4);
  float* kbuf = (float*)take((size_t)NPAD * RANKX * 4);
  _Float16* E16 = (_Float16*)take((size_t)VOCABX * DIMX * 2);
  _Float16* Wp16 = (_Float16*)take((size_t)DIMX * DIMX * 2);
  _Float16* Wqt = (_Float16*)take((size_t)2 * RANKX * DIMX * 2);
  _Float16* Wkt = (_Float16*)take((size_t)2 * RANKX * DIMX * 2);
  float* M0 = (float*)take((size_t)SLOT0 * DIMX * 4);
  float* M1 = (float*)take((size_t)SLOT1 * DIMX * 4);
  float* M2 = (float*)take((size_t)SLOT2 * DIMX * 4);
  float* KR0 = (float*)take((size_t)SLOT0 * RANKX * 4);
  float* KR1 = (float*)take((size_t)SLOT1 * RANKX * 4);
  float* KR2 = (float*)take((size_t)SLOT2 * RANKX * 4);

  // 1) embedding gather (shifted by one token)
  embed_k<<<(NSEQ * DIMX + 255) / 256, 256, 0, stream>>>(ids, E, s);

  // 2) weight conversions for WMMA
  cvt_k<<<(VOCABX * DIMX + 255) / 256, 256, 0, stream>>>(E, E16, VOCABX * DIMX,
                                                         VOCABX * DIMX);
  cvt_k<<<(DIMX * DIMX + 255) / 256, 256, 0, stream>>>(Wproj, Wp16, DIMX * DIMX,
                                                       DIMX * DIMX);
  tw_k<<<(2 * RANKX * DIMX + 255) / 256, 256, 0, stream>>>(Wq, Wqt);
  tw_k<<<(2 * RANKX * DIMX + 255) / 256, 256, 0, stream>>>(Wk, Wkt);

  // 3) sequential hierarchical memory scan -> q (f32 + f16)
  scan_k<<<1, 512, 0, stream>>>(s, Ur, Vr, m00, m01, m02, M0, M1, M2, KR0, KR1,
                                KR2, q, qf16);

  // 4) h = LN(q @ Wproj^T)
  {
    int tilesM = NPAD / 16, tilesN = DIMX / 64;
    gemm_k<<<tilesM * tilesN / 8, 256, 0, stream>>>(qf16, Wp16, tmp, NSEQ, DIMX,
                                                    DIMX, tilesN, tilesM);
    ln_k<<<NSEQ, 256, 0, stream>>>(tmp, ln_in_g, ln_in_b, h);
  }

  // 5) two players of signed windowed attention
  for (int p = 0; p < 2; ++p) {
    cvt_k<<<(NPAD * DIMX + 255) / 256, 256, 0, stream>>>(h, hf16, NPAD * DIMX,
                                                         NSEQ * DIMX);
    int tilesM = NPAD / 16, tilesN = RANKX / 64;  // tilesN = 1
    gemm_k<<<tilesM * tilesN / 8, 256, 0, stream>>>(
        hf16, Wqt + (size_t)p * RANKX * DIMX, abuf, NSEQ, DIMX, RANKX, tilesN,
        tilesM);
    gemm_k<<<tilesM * tilesN / 8, 256, 0, stream>>>(
        hf16, Wkt + (size_t)p * RANKX * DIMX, kbuf, NSEQ, DIMX, RANKX, tilesN,
        tilesM);
    attn_k<<<NSEQ, 64, 0, stream>>>(h, abuf, kbuf, tmp);
    ln_k<<<NSEQ, 256, 0, stream>>>(tmp, ln_pred_g + (size_t)p * DIMX,
                                   ln_pred_b + (size_t)p * DIMX, h);
  }

  // 6) out = LN(h); logits = out @ E^T  (dominant WMMA GEMM: 2047x512x32000)
  ln_k<<<NSEQ, 256, 0, stream>>>(h, ln_out_g, ln_out_b, tmp);
  cvt_k<<<(NPAD * DIMX + 255) / 256, 256, 0, stream>>>(tmp, qf16, NPAD * DIMX,
                                                       NSEQ * DIMX);
  {
    // 64x64 wave tiles: 32 M-supertiles x 500 N-tiles, 4 waves per block
    int ngrp = (VOCABX / 64) / 4;  // 125
    gemm64_k<<<(NPAD / 64) * ngrp, 128, 0, stream>>>(qf16, E16, out, NSEQ,
                                                     VOCABX, ngrp);
  }
}
